// TensorProductConvLayer_7928509628864
// MI455X (gfx1250) — matrix operand order repaired and checked by reading, not compiled, not run on hardware
//
#include <hip/hip_runtime.h>
#include <hip/hip_bf16.h>

typedef __attribute__((ext_vector_type(16))) _Float16 v16h;
typedef __attribute__((ext_vector_type(8)))  float    v8f;

#define N_NODES   8192
#define N_EDGES   65536
#define EDGE_FEAT 64
#define HIDDEN    64
#define FEAT      136
#define WNUM      3392
#define NT_W2     212   // 3392 / 16 column tiles
#define NT_W1     4     // 64 / 16
#define KT        2     // 64 / 32 K tiles

// D = A(16x32 f16) * B(32x16 f16) + C(16x16 f32)
__device__ __forceinline__ v8f wmma_f16(v16h a, v16h b, v8f c) {
  return __builtin_amdgcn_wmma_f32_16x16x32_f16(false, a, false, b, (short)0, c, false, false);
}

// Pack a row-major f32 KxN weight matrix into per-lane WMMA B-operand layout, f16.
// Layout: [ntile][ktile][lane(32)][j(16)] ; B operand: lane holds column n = tile*16 + (lane&15),
// K = kt*32 + (j<8 ? (lane>>4)*8 + j : 16 + (lane>>4)*8 + (j-8)).
__global__ void pack_b_f16(const float* __restrict__ w, int N, int ktiles,
                           _Float16* __restrict__ out, int total) {
  int idx = blockIdx.x * blockDim.x + threadIdx.x;
  if (idx >= total) return;
  int j    = idx & 15;
  int lane = (idx >> 4) & 31;
  int kt   = (idx >> 9) % ktiles;
  int tile = idx / (512 * ktiles);
  int k = kt * 32 + (j < 8 ? (lane >> 4) * 8 + j : 16 + (lane >> 4) * 8 + (j - 8));
  int n = tile * 16 + (lane & 15);
  out[idx] = (_Float16)w[(size_t)k * N + n];
}

__global__ void zero_f32(float* __restrict__ p, int n) {
  int i = blockIdx.x * blockDim.x + threadIdx.x;
  if (i < n) p[i] = 0.f;
}

__global__ void finalize_k(const float* __restrict__ nsum, const float* __restrict__ ncnt,
                           const float* __restrict__ node_attr, float* __restrict__ out, int total) {
  int i = blockIdx.x * blockDim.x + threadIdx.x;
  if (i >= total) return;
  int n = i / FEAT;
  float c = ncnt[n];
  c = c > 1.f ? c : 1.f;
  out[i] = nsum[i] / c + node_attr[i];
}

// One block = 16 edges, 256 threads = 8 waves.
__launch_bounds__(256, 1)
__global__ void tpconv_fused(const float* __restrict__ node_attr,
                             const int*   __restrict__ edge_index,
                             const float* __restrict__ edge_attr,
                             const float* __restrict__ edge_sh,
                             const _Float16* __restrict__ w1p,
                             const float* __restrict__ fc_b1,
                             const _Float16* __restrict__ w2p,
                             const float* __restrict__ fc_b2,
                             float* __restrict__ nsum,
                             float* __restrict__ ncnt)
{
  __shared__ float hid[16][HIDDEN];      // relu(FC1) per edge
  __shared__ float s_o0e[16][48];
  __shared__ float s_o1o[16][64][3];
  __shared__ float s_o1e[16][40][3];
  __shared__ float s_o0o[16][24];
  __shared__ float yl[16][FEAT];         // per-edge tp outputs (cross-wave reduced)
  __shared__ int   srcIdx[16];

  const int tid  = threadIdx.x;
  const int w    = tid >> 5;
  const int lane = tid & 31;
  const int nl   = lane & 15;
  const int hi   = lane >> 4;
  const int e0   = blockIdx.x * 16;

  if (w < 4) {
    // ---- FC1 via WMMA: this wave produces hidden columns [w*16, w*16+16) for all 16 edges.
    // A operand (16x32 f16): lane holds row m = nl; K pattern per ISA: j<8 -> hi*8+j, j>=8 -> 16+hi*8+(j-8).
    const float* row = edge_attr + (size_t)(e0 + nl) * EDGE_FEAT;
    v16h a0, a1;
    #pragma unroll
    for (int j = 0; j < 8; ++j) {
      a0[j]     = (_Float16)row[hi * 8 + j];
      a0[j + 8] = (_Float16)row[16 + hi * 8 + j];
      a1[j]     = (_Float16)row[32 + hi * 8 + j];
      a1[j + 8] = (_Float16)row[48 + hi * 8 + j];
    }
    v16h b0 = *(const v16h*)(w1p + ((size_t)(w * KT + 0) * 32 + lane) * 16);
    v16h b1 = *(const v16h*)(w1p + ((size_t)(w * KT + 1) * 32 + lane) * 16);
    float bias = fc_b1[w * 16 + nl];
    v8f c;
    #pragma unroll
    for (int r = 0; r < 8; ++r) c[r] = bias;
    c = wmma_f16(a0, b0, c);
    c = wmma_f16(a1, b1, c);
    #pragma unroll
    for (int r = 0; r < 8; ++r) {           // C layout: row = r + hi*8, col = nl
      float v = c[r];
      hid[r + hi * 8][w * 16 + nl] = v > 0.f ? v : 0.f;
    }
  } else {
    // ---- o-feature construction (gather node_attr[dst], combine with spherical harmonics)
    const int t2 = tid - 128;  // 0..127: 8 workers per edge
    const int e  = t2 >> 3;
    const int s  = t2 & 7;
    const int ge = e0 + e;
    if (s == 0) srcIdx[e] = edge_index[ge];
    const int dst = edge_index[N_EDGES + ge];
    const float* x = node_attr + (size_t)dst * FEAT;
    const float sh0 = edge_sh[ge * 4 + 0];
    const float shx = edge_sh[ge * 4 + 1];
    const float shy = edge_sh[ge * 4 + 2];
    const float shz = edge_sh[ge * 4 + 3];
    const float INV_S3 = 0.57735026918962576f;  // 1/sqrt(3)
    const float INV_S2 = 0.70710678118654752f;  // 1/sqrt(2)
    for (int t = s; t < 48; t += 8) {           // o0e = [x0e*sh0 ; (x1o . sh1)/sqrt3]
      float v;
      if (t < 32) v = x[t] * sh0;
      else { const float* u = x + 32 + (t - 32) * 3;
             v = (u[0] * shx + u[1] * shy + u[2] * shz) * INV_S3; }
      s_o0e[e][t] = v;
    }
    for (int t = s; t < 64; t += 8) {           // o1o = [x0e (x) sh1 ; x1o*sh0 ; cross(x1e,sh1)/sqrt2]
      float vx, vy, vz;
      if (t < 32) { float a = x[t]; vx = a * shx; vy = a * shy; vz = a * shz; }
      else if (t < 48) { const float* u = x + 32 + (t - 32) * 3;
                         vx = u[0] * sh0; vy = u[1] * sh0; vz = u[2] * sh0; }
      else { const float* u = x + 80 + (t - 48) * 3;
             vx = (u[1] * shz - u[2] * shy) * INV_S2;
             vy = (u[2] * shx - u[0] * shz) * INV_S2;
             vz = (u[0] * shy - u[1] * shx) * INV_S2; }
      s_o1o[e][t][0] = vx; s_o1o[e][t][1] = vy; s_o1o[e][t][2] = vz;
    }
    for (int t = s; t < 40; t += 8) {           // o1e = [cross(x1o,sh1)/sqrt2 ; x1e*sh0 ; x0o (x) sh1]
      float vx, vy, vz;
      if (t < 16) { const float* u = x + 32 + t * 3;
                    vx = (u[1] * shz - u[2] * shy) * INV_S2;
                    vy = (u[2] * shx - u[0] * shz) * INV_S2;
                    vz = (u[0] * shy - u[1] * shx) * INV_S2; }
      else if (t < 32) { const float* u = x + 80 + (t - 16) * 3;
                         vx = u[0] * sh0; vy = u[1] * sh0; vz = u[2] * sh0; }
      else { float a = x[128 + (t - 32)]; vx = a * shx; vy = a * shy; vz = a * shz; }
      s_o1e[e][t][0] = vx; s_o1e[e][t][1] = vy; s_o1e[e][t][2] = vz;
    }
    for (int t = s; t < 24; t += 8) {           // o0o = [(x1e . sh1)/sqrt3 ; x0o*sh0]
      float v;
      if (t < 16) { const float* u = x + 80 + t * 3;
                    v = (u[0] * shx + u[1] * shy + u[2] * shz) * INV_S3; }
      else v = x[128 + (t - 16)] * sh0;
      s_o0o[e][t] = v;
    }
    for (int i = t2; i < 16 * FEAT; i += 128) ((float*)yl)[i] = 0.f;
  }
  __syncthreads();

  // ---- FC2 A operand (hidden -> f16), shared by all waves of this block.
  v16h A0, A1;
  {
    const float* hrow = hid[nl];
    #pragma unroll
    for (int j = 0; j < 8; ++j) {
      A0[j]     = (_Float16)hrow[hi * 8 + j];
      A0[j + 8] = (_Float16)hrow[16 + hi * 8 + j];
      A1[j]     = (_Float16)hrow[32 + hi * 8 + j];
      A1[j + 8] = (_Float16)hrow[48 + hi * 8 + j];
    }
  }

  // Per-lane output partials (edge = r + hi*8, column tied to nl by segment mapping)
  v8f y0e_a0 = {}, y0e_a1 = {};
  v8f y1o_a0 = {}, y1o_a1 = {}, y1o_a2 = {};
  v8f y1e_a0 = {}, y1e_a1 = {}, y1e_a2 = {};
  v8f y0o_a  = {};

  // Stream W column tiles: 2 WMMAs (K=64) per tile, contract immediately.
  for (int t = w; t < NT_W2; t += 8) {
    v16h b0 = *(const v16h*)(w2p + ((size_t)(t * KT + 0) * 32 + lane) * 16);
    v16h b1 = *(const v16h*)(w2p + ((size_t)(t * KT + 1) * 32 + lane) * 16);
    float bias = fc_b2[t * 16 + nl];
    v8f c;
    #pragma unroll
    for (int r = 0; r < 8; ++r) c[r] = bias;
    c = wmma_f16(A0, b0, c);
    c = wmma_f16(A1, b1, c);

    if (t < 96) {                       // segment 0e: k = i*32 + o, tile -> (i = t>>1, o half = t&1)
      const int i = t >> 1;
      if (t & 1) {
        #pragma unroll
        for (int r = 0; r < 8; ++r) y0e_a1[r] += c[r] * s_o0e[r + hi * 8][i];
      } else {
        #pragma unroll
        for (int r = 0; r < 8; ++r) y0e_a0[r] += c[r] * s_o0e[r + hi * 8][i];
      }
    } else if (t < 160) {               // segment 1o: k = i*16 + o, i = t-96, o = nl, 3 components
      const int i = t - 96;
      #pragma unroll
      for (int r = 0; r < 8; ++r) {
        const float* o = s_o1o[r + hi * 8][i];
        float wv = c[r];
        y1o_a0[r] += wv * o[0];
        y1o_a1[r] += wv * o[1];
        y1o_a2[r] += wv * o[2];
      }
    } else if (t < 200) {               // segment 1e: i = t-160, o = nl
      const int i = t - 160;
      #pragma unroll
      for (int r = 0; r < 8; ++r) {
        const float* o = s_o1e[r + hi * 8][i];
        float wv = c[r];
        y1e_a0[r] += wv * o[0];
        y1e_a1[r] += wv * o[1];
        y1e_a2[r] += wv * o[2];
      }
    } else {                            // segment 0o: k = i*8 + o; tile covers i = (t-200)*2 + (nl>>3)
      const int i = (t - 200) * 2 + (nl >> 3);
      #pragma unroll
      for (int r = 0; r < 8; ++r) y0o_a[r] += c[r] * s_o0o[r + hi * 8][i];
    }
  }

  // Cross-wave reduction into yl with segment normalizations 1/sqrt(i_dim).
  const float S0E = 0.14433756729740643f;   // 1/sqrt(48)
  const float S1O = 0.125f;                 // 1/sqrt(64)
  const float S1E = 0.15811388300841897f;   // 1/sqrt(40)
  const float S0O = 0.20412414523193150f;   // 1/sqrt(24)
  #pragma unroll
  for (int r = 0; r < 8; ++r) {
    const int e = r + hi * 8;
    atomicAdd(&yl[e][nl],               y0e_a0[r] * S0E);
    atomicAdd(&yl[e][16 + nl],          y0e_a1[r] * S0E);
    atomicAdd(&yl[e][32 + nl * 3 + 0],  y1o_a0[r] * S1O);   // y1o flat: o*3 + c
    atomicAdd(&yl[e][32 + nl * 3 + 1],  y1o_a1[r] * S1O);
    atomicAdd(&yl[e][32 + nl * 3 + 2],  y1o_a2[r] * S1O);
    atomicAdd(&yl[e][80 + nl * 3 + 0],  y1e_a0[r] * S1E);
    atomicAdd(&yl[e][80 + nl * 3 + 1],  y1e_a1[r] * S1E);
    atomicAdd(&yl[e][80 + nl * 3 + 2],  y1e_a2[r] * S1E);
    atomicAdd(&yl[e][128 + (nl & 7)],   y0o_a[r]  * S0O);
  }
  __syncthreads();

  // Scatter into node accumulators (L2-resident, 4.5 MB).
  for (int i = tid; i < 16 * FEAT; i += 256) {
    const int e = i / FEAT;
    const int f = i - e * FEAT;
    atomicAdd(&nsum[(size_t)srcIdx[e] * FEAT + f], yl[e][f]);
  }
  if (tid < 16) atomicAdd(&ncnt[srcIdx[tid]], 1.0f);
}

extern "C" void kernel_launch(void* const* d_in, const int* in_sizes, int n_in,
                              void* d_out, int out_size, void* d_ws, size_t ws_size,
                              hipStream_t stream) {
  const float* node_attr = (const float*)d_in[0];
  const int*   edge_index= (const int*)d_in[1];
  const float* edge_attr = (const float*)d_in[2];
  const float* edge_sh   = (const float*)d_in[3];
  const float* fc_w1     = (const float*)d_in[4];
  const float* fc_b1     = (const float*)d_in[5];
  const float* fc_w2     = (const float*)d_in[6];
  const float* fc_b2     = (const float*)d_in[7];
  float* out = (float*)d_out;
  (void)in_sizes; (void)n_in; (void)out_size; (void)ws_size;

  char* ws = (char*)d_ws;
  size_t off = 0;
  auto alloc = [&](size_t bytes) -> void* {
    void* p = ws + off;
    off = (off + bytes + 255) & ~(size_t)255;
    return p;
  };
  _Float16* w1p = (_Float16*)alloc((size_t)NT_W1 * KT * 512 * sizeof(_Float16));
  _Float16* w2p = (_Float16*)alloc((size_t)NT_W2 * KT * 512 * sizeof(_Float16));
  float* nsum   = (float*)alloc((size_t)N_NODES * FEAT * sizeof(float));
  float* ncnt   = (float*)alloc((size_t)N_NODES * sizeof(float));

  {
    int tot = NT_W1 * KT * 512;
    pack_b_f16<<<(tot + 255) / 256, 256, 0, stream>>>(fc_w1, HIDDEN, KT, w1p, tot);
  }
  {
    int tot = NT_W2 * KT * 512;
    pack_b_f16<<<(tot + 255) / 256, 256, 0, stream>>>(fc_w2, WNUM, KT, w2p, tot);
  }
  zero_f32<<<(N_NODES * FEAT + 255) / 256, 256, 0, stream>>>(nsum, N_NODES * FEAT);
  zero_f32<<<(N_NODES + 255) / 256, 256, 0, stream>>>(ncnt, N_NODES);

  tpconv_fused<<<N_EDGES / 16, 256, 0, stream>>>(node_attr, edge_index, edge_attr, edge_sh,
                                                 w1p, fc_b1, w2p, fc_b2, nsum, ncnt);

  finalize_k<<<(N_NODES * FEAT + 255) / 256, 256, 0, stream>>>(nsum, ncnt, node_attr, out,
                                                               N_NODES * FEAT);
}
